// Decoder_36696200577047
// MI455X (gfx1250) — compile-verified, matrix operand order checked by reading, not run on hardware
//
#include <hip/hip_runtime.h>
#include <hip/hip_bf16.h>

// MI455X / gfx1250 decoder step. All GEMMs via v_wmma_f32_16x16x32_bf16
// (f32 inputs converted to bf16 in-register; f32 accumulate).
// 128x64 macro-tile, 8 waves, 4 WMMA tiles/wave/step, double-buffered LDS.
// f32->bf16: round-to-nearest + v_perm_b32 pair-pack (3 VALU per 2 elements).
// B tile packs along k so the col-major LDS transpose uses aligned b32 stores.

typedef __attribute__((ext_vector_type(16))) __bf16 v16bf;
typedef __attribute__((ext_vector_type(8)))  float  v8f;

#define U_  1024   // UNITS
#define EM_ 512    // EMBED
#define EN_ 1024   // ENC
#define B_  256    // batch
#define T_  128    // timesteps
#define VOC 32000

union Frag { uint4 u[2]; v16bf v; };

// Pack two f32 into one dword of 2x bf16 (round-to-nearest).
__device__ __forceinline__ unsigned pack2bf(float a, float b) {
  union { float f; unsigned u; } x, y; x.f = a; y.f = b;
  unsigned ra = x.u + 0x8000u;
  unsigned rb = y.u + 0x8000u;
#if __has_builtin(__builtin_amdgcn_perm)
  return __builtin_amdgcn_perm(rb, ra, 0x07060302u);  // {rb[31:16], ra[31:16]}
#else
  return (rb & 0xFFFF0000u) | (ra >> 16);
#endif
}

__device__ __forceinline__ void cvt8_store(float4 f0, float4 f1, unsigned short* d) {
  uint4 q;
  q.x = pack2bf(f0.x, f0.y);
  q.y = pack2bf(f0.z, f0.w);
  q.z = pack2bf(f1.x, f1.y);
  q.w = pack2bf(f1.z, f1.w);
  *(uint4*)d = q;                                   // 16B-aligned LDS store
}

// ---- register staging for double buffering -----------------------------------------
struct StageA { float4 f[4]; };   // 16 consecutive f32 of one A row
struct StageB { float4 f[2]; };   // 2 (k) x 4 (n) f32 sub-block of B

// A tile: 128 rows x 32 k (row-major, stride lda). Thread t -> row t>>1, col (t&1)*16.
__device__ __forceinline__ StageA ldgA(const float* A, long lda, int row0, int k0) {
  int t = threadIdx.x;
  const float* s = A + (long)(row0 + (t >> 1)) * lda + (k0 + (t & 1) * 16);
  StageA sa;
  sa.f[0] = *(const float4*)(s);
  sa.f[1] = *(const float4*)(s + 4);
  sa.f[2] = *(const float4*)(s + 8);
  sa.f[3] = *(const float4*)(s + 12);
  return sa;
}
__device__ __forceinline__ void stA(const StageA& sa, unsigned short* lds) {
  int t = threadIdx.x;
  unsigned short* d = lds + (t >> 1) * 32 + (t & 1) * 16;
  cvt8_store(sa.f[0], sa.f[1], d);
  cvt8_store(sa.f[2], sa.f[3], d + 8);
}

// B tile: 32 k x 64 n (row-major, stride ldb) -> LDS col-major [64 cols][32 k].
// Thread t owns k-pair k=2*(t>>4), n-quad n=4*(t&15): 16 consecutive threads read one
// contiguous 256B row (coalesced); packing along k makes each store an aligned b32.
__device__ __forceinline__ StageB ldgB(const float* Bm, long ldb, int k0, int n0) {
  int t = threadIdx.x;
  const float* s = Bm + (long)(k0 + (t >> 4) * 2) * ldb + (n0 + (t & 15) * 4);
  StageB sb;
  sb.f[0] = *(const float4*)(s);          // row k
  sb.f[1] = *(const float4*)(s + ldb);    // row k+1
  return sb;
}
__device__ __forceinline__ void stB(const StageB& sb, unsigned short* lds) {
  int t = threadIdx.x;
  int k = (t >> 4) * 2;
  int n = (t & 15) * 4;
  unsigned* base = (unsigned*)lds;        // dword view of [64][32] ushort tile
  base[((n + 0) * 32 + k) >> 1] = pack2bf(sb.f[0].x, sb.f[1].x);
  base[((n + 1) * 32 + k) >> 1] = pack2bf(sb.f[0].y, sb.f[1].y);
  base[((n + 2) * 32 + k) >> 1] = pack2bf(sb.f[0].z, sb.f[1].z);
  base[((n + 3) * 32 + k) >> 1] = pack2bf(sb.f[0].w, sb.f[1].w);
}

// Read one 16x32 bf16 fragment from LDS tile [rows][32]; idx = row (A) or col (B)
__device__ __forceinline__ v16bf read_frag(const unsigned short* lds, int idx, int kb) {
  const uint4* p = (const uint4*)(lds + idx * 32 + kb);
  Frag f; f.u[0] = p[0]; f.u[1] = p[2];             // K runs kb..kb+7, kb+16..kb+23
  return f.v;
}

__device__ __forceinline__ v8f wmma_bf16(v16bf a, v16bf b, v8f c) {
  return __builtin_amdgcn_wmma_f32_16x16x32_bf16(false, a, false, b, (short)0, c,
                                                 false, false);
}

// A-source functors --------------------------------------------------------------------
struct APlain {
  const float* p; long ld;
  __device__ StageA operator()(int row0, int k0) const { return ldgA(p, ld, row0, k0); }
};
struct AXt {   // xt = [context(1024) | E[x](512)]; 16-runs never straddle k=1024
  const float* ctx; const float* E; const int* x;
  __device__ StageA operator()(int row0, int k0) const {
    int t = threadIdx.x;
    int row = row0 + (t >> 1);
    int k   = k0 + (t & 1) * 16;
    const float* s = (k < EN_) ? (ctx + (long)row * EN_ + k)
                               : (E + (long)x[row] * EM_ + (k - EN_));
    StageA sa;
    sa.f[0] = *(const float4*)(s);
    sa.f[1] = *(const float4*)(s + 4);
    sa.f[2] = *(const float4*)(s + 8);
    sa.f[3] = *(const float4*)(s + 12);
    return sa;
  }
};

// Core: 128x64 output tile, K-loop with double-buffered LDS, 4 WMMA per wave per step.
template <class AL>
__device__ __forceinline__ void gemm128x64(const AL& al, const float* Bm, long ldb,
                                           int row0, int n0, int K, v8f cc[2][2]) {
  __shared__ unsigned short Asl[2][128 * 32];
  __shared__ unsigned short Bsl[2][64 * 32];
  const int w    = threadIdx.x >> 5;
  const int rt0  = (w >> 1) * 2;
  const int ct0  = (w & 1) * 2;
  const int lane = threadIdx.x & 31;
  const int l16  = lane & 15;
  const int kb   = (lane >> 4) * 8;

  StageA sa = al(row0, 0);
  StageB sb = ldgB(Bm, ldb, 0, n0);
  stA(sa, Asl[0]);
  stB(sb, Bsl[0]);
  __syncthreads();

  const int NS = K / 32;
  for (int s = 0; s < NS; ++s) {
    const int cur = s & 1;
    if (s + 1 < NS) {                       // issue next tile's global loads early:
      sa = al(row0, (s + 1) * 32);          // VMEM latency hides under the WMMAs
      sb = ldgB(Bm, ldb, (s + 1) * 32, n0);
    }
    v16bf a0 = read_frag(Asl[cur], (rt0    ) * 16 + l16, kb);
    v16bf a1 = read_frag(Asl[cur], (rt0 + 1) * 16 + l16, kb);
    v16bf b0 = read_frag(Bsl[cur], (ct0    ) * 16 + l16, kb);
    v16bf b1 = read_frag(Bsl[cur], (ct0 + 1) * 16 + l16, kb);
    cc[0][0] = wmma_bf16(a0, b0, cc[0][0]);
    cc[0][1] = wmma_bf16(a0, b1, cc[0][1]);
    cc[1][0] = wmma_bf16(a1, b0, cc[1][0]);
    cc[1][1] = wmma_bf16(a1, b1, cc[1][1]);
    if (s + 1 < NS) {                       // write other buffer; reads of it are
      stA(sa, Asl[1 - cur]);                // separated by the single barrier below
      stB(sb, Bsl[1 - cur]);
    }
    __syncthreads();
  }
}

#define TILE_IDX() \
  const int w    = threadIdx.x >> 5; \
  const int rt0  = (w >> 1) * 2; \
  const int ct0  = (w & 1) * 2; \
  const int half = (threadIdx.x & 31) >> 4; \
  const int l16  = threadIdx.x & 15;

#define ZERO_CC() \
  v8f cc[2][2]; \
  _Pragma("unroll") for (int i = 0; i < 2; ++i) \
  _Pragma("unroll") for (int j = 0; j < 2; ++j) \
    cc[i][j] = (v8f){0.f,0.f,0.f,0.f,0.f,0.f,0.f,0.f};

// ---------------- Kernel 1: hproj = hidden @ W2 + b1 + b2  (256x1024, K=1024) --------
__global__ __launch_bounds__(256) void k_hproj(const float* __restrict__ hidden,
                                               const float* __restrict__ W2,
                                               const float* __restrict__ b1,
                                               const float* __restrict__ b2,
                                               float* __restrict__ hproj) {
  ZERO_CC();
  const int row0 = blockIdx.x * 128;
  const int n0   = blockIdx.y * 64;
  APlain al{hidden, U_};
  gemm128x64(al, W2, U_, row0, n0, U_, cc);
  TILE_IDX();
#pragma unroll
  for (int j = 0; j < 2; ++j) {
    int n = n0 + (ct0 + j) * 16 + l16;
    float bb = b1[n] + b2[n];
#pragma unroll
    for (int i = 0; i < 2; ++i)
#pragma unroll
      for (int vi = 0; vi < 8; ++vi) {
        int m = row0 + (rt0 + i) * 16 + half * 8 + vi;
        hproj[m * U_ + n] = cc[i][j][vi] + bb;
      }
  }
}

// -------- Kernel 2: score[b,t] += sum_n tanh((enc@W1)[bt,n] + hproj[b,n]) * V[n] -----
__global__ __launch_bounds__(256) void k_score(const float* __restrict__ enc,
                                               const float* __restrict__ W1,
                                               const float* __restrict__ hproj,
                                               const float* __restrict__ V,
                                               float* __restrict__ score) {
  ZERO_CC();
  const int b    = blockIdx.x;               // 128 rows == one batch element
  const int row0 = b * T_;
  const int n0   = blockIdx.y * 64;
  APlain al{enc, EN_};
  gemm128x64(al, W1, U_, row0, n0, EN_, cc);
  TILE_IDX();
  float part[2][8];
#pragma unroll
  for (int i = 0; i < 2; ++i)
#pragma unroll
    for (int vi = 0; vi < 8; ++vi) part[i][vi] = 0.f;
#pragma unroll
  for (int j = 0; j < 2; ++j) {
    int n = n0 + (ct0 + j) * 16 + l16;
    float hp = hproj[b * U_ + n];
    float vv = V[n];
#pragma unroll
    for (int i = 0; i < 2; ++i)
#pragma unroll
      for (int vi = 0; vi < 8; ++vi)
        part[i][vi] += tanhf(cc[i][j][vi] + hp) * vv;
  }
  // reduce across the 16 lanes holding different n (xor masks stay within halves)
#pragma unroll
  for (int mask = 8; mask >= 1; mask >>= 1)
#pragma unroll
    for (int i = 0; i < 2; ++i)
#pragma unroll
      for (int vi = 0; vi < 8; ++vi)
        part[i][vi] += __shfl_xor(part[i][vi], mask);
  if (l16 == 0) {
#pragma unroll
    for (int i = 0; i < 2; ++i)
#pragma unroll
      for (int vi = 0; vi < 8; ++vi)
        atomicAdd(&score[row0 + (rt0 + i) * 16 + half * 8 + vi], part[i][vi]);
  }
}

// ---------------- Kernel 3: softmax over T + context = sum_t attn*enc ----------------
__global__ __launch_bounds__(256) void k_softmax_ctx(const float* __restrict__ score,
                                                     const float* __restrict__ enc,
                                                     float* __restrict__ attn,
                                                     float* __restrict__ ctx) {
  const int b = blockIdx.x;
  const int t = threadIdx.x;
  __shared__ float red[256];
  __shared__ float sattn[T_];
  float v = (t < T_) ? score[b * T_ + t] : -3.0e38f;
  red[t] = v; __syncthreads();
  for (int s = 128; s > 0; s >>= 1) { if (t < s) red[t] = fmaxf(red[t], red[t + s]); __syncthreads(); }
  float mx = red[0]; __syncthreads();
  float e = (t < T_) ? expf(v - mx) : 0.f;
  red[t] = e; __syncthreads();
  for (int s = 128; s > 0; s >>= 1) { if (t < s) red[t] += red[t + s]; __syncthreads(); }
  float inv = 1.f / red[0];
  if (t < T_) { float a = e * inv; sattn[t] = a; attn[b * T_ + t] = a; }
  __syncthreads();
  // context: each thread owns 4 consecutive enc columns
  float4 acc = {0.f, 0.f, 0.f, 0.f};
  const float* ebase = enc + (long)b * T_ * EN_ + t * 4;
  for (int tt = 0; tt < T_; ++tt) {
    float a = sattn[tt];
    float4 ev = *(const float4*)(ebase + (long)tt * EN_);
    acc.x += a * ev.x; acc.y += a * ev.y; acc.z += a * ev.z; acc.w += a * ev.w;
  }
  *(float4*)(ctx + b * EN_ + t * 4) = acc;
}

// ---------------- Kernel 4: gx = [context, E[x]] @ Wx + b_gru  (skip r gate) ---------
__global__ __launch_bounds__(256) void k_gx(const float* __restrict__ ctx,
                                            const float* __restrict__ E,
                                            const int* __restrict__ x,
                                            const float* __restrict__ Wx,
                                            const float* __restrict__ bg,
                                            float* __restrict__ gx) {
  ZERO_CC();
  const int row0 = blockIdx.x * 128;
  const int cb   = blockIdx.y;                          // 0..31
  const int n0   = (cb < 16) ? cb * 64 : 2048 + (cb - 16) * 64;   // z cols / hh cols
  AXt al{ctx, E, x};
  gemm128x64(al, Wx, 3 * U_, row0, n0, EN_ + EM_, cc);
  TILE_IDX();
#pragma unroll
  for (int j = 0; j < 2; ++j) {
    int n = n0 + (ct0 + j) * 16 + l16;
    float bb = bg[n];
#pragma unroll
    for (int i = 0; i < 2; ++i)
#pragma unroll
      for (int vi = 0; vi < 8; ++vi) {
        int m = row0 + (rt0 + i) * 16 + half * 8 + vi;
        gx[(long)m * (3 * U_) + n] = cc[i][j][vi] + bb;
      }
  }
}

// ---------------- Kernel 5: state = (1 - sigmoid(z)) * tanh(hh) ----------------------
__global__ __launch_bounds__(256) void k_state(const float* __restrict__ gx,
                                               float* __restrict__ state) {
  int i = blockIdx.x * blockDim.x + threadIdx.x;        // 0 .. 256*1024-1
  int b = i >> 10, u = i & (U_ - 1);
  float z  = gx[(long)b * (3 * U_) + u];
  float hh = gx[(long)b * (3 * U_) + 2 * U_ + u];
  float sz = 1.f / (1.f + expf(-z));
  state[i] = (1.f - sz) * tanhf(hh);
}

// ---------------- Kernel 6: logits = state @ Wfc + bfc  (256x32000, K=1024) ----------
__global__ __launch_bounds__(256) void k_logits(const float* __restrict__ state,
                                                const float* __restrict__ Wfc,
                                                const float* __restrict__ bfc,
                                                float* __restrict__ logits) {
  ZERO_CC();
  const int row0 = blockIdx.x * 128;
  const int n0   = blockIdx.y * 64;
  APlain al{state, U_};
  gemm128x64(al, Wfc, VOC, row0, n0, U_, cc);
  TILE_IDX();
#pragma unroll
  for (int j = 0; j < 2; ++j) {
    int n = n0 + (ct0 + j) * 16 + l16;
    float bb = bfc[n];
#pragma unroll
    for (int i = 0; i < 2; ++i)
#pragma unroll
      for (int vi = 0; vi < 8; ++vi) {
        int m = row0 + (rt0 + i) * 16 + half * 8 + vi;
        logits[(long)m * VOC + n] = cc[i][j][vi] + bb;
      }
  }
}

// -------------------------------------------------------------------------------------
extern "C" void kernel_launch(void* const* d_in, const int* in_sizes, int n_in,
                              void* d_out, int out_size, void* d_ws, size_t ws_size,
                              hipStream_t stream) {
  const int*   x      = (const int*)  d_in[0];
  const float* hidden = (const float*)d_in[1];
  const float* enc    = (const float*)d_in[2];
  const float* E      = (const float*)d_in[3];
  const float* W1     = (const float*)d_in[4];
  const float* b1     = (const float*)d_in[5];
  const float* W2     = (const float*)d_in[6];
  const float* b2     = (const float*)d_in[7];
  const float* V      = (const float*)d_in[8];
  // d_in[9] = bv: softmax is shift-invariant and raw score is not returned -> unused
  const float* Wx     = (const float*)d_in[10];
  // d_in[11] = Wh: unused by reference
  const float* bg     = (const float*)d_in[12];
  const float* Wfc    = (const float*)d_in[13];
  const float* bfc    = (const float*)d_in[14];
  (void)in_sizes; (void)n_in; (void)out_size; (void)ws_size;

  float* out    = (float*)d_out;
  float* logits = out;                       // 256*32000
  float* state  = out + (long)B_ * VOC;      // 256*1024
  float* attn   = state + (long)B_ * U_;     // 256*128

  float* ws    = (float*)d_ws;
  float* hproj = ws;                         // 256*1024
  float* score = hproj + (long)B_ * U_;      // 256*128
  float* ctx   = score + (long)B_ * T_;      // 256*1024
  float* gx    = ctx + (long)B_ * U_;        // 256*3072

  hipMemsetAsync(score, 0, (size_t)B_ * T_ * sizeof(float), stream);

  hipLaunchKernelGGL(k_hproj,       dim3(2, 16),   dim3(256), 0, stream,
                     hidden, W2, b1, b2, hproj);
  hipLaunchKernelGGL(k_score,       dim3(256, 16), dim3(256), 0, stream,
                     enc, W1, hproj, V, score);
  hipLaunchKernelGGL(k_softmax_ctx, dim3(256),     dim3(256), 0, stream,
                     score, enc, attn, ctx);
  hipLaunchKernelGGL(k_gx,          dim3(2, 32),   dim3(256), 0, stream,
                     ctx, E, x, Wx, bg, gx);
  hipLaunchKernelGGL(k_state,       dim3(1024),    dim3(256), 0, stream,
                     gx, state);
  hipLaunchKernelGGL(k_logits,      dim3(2, 500),  dim3(256), 0, stream,
                     state, Wfc, bfc, logits);
}
